// VQVAE_37245956391381
// MI455X (gfx1250) — matrix operand order repaired and checked by reading, not compiled, not run on hardware
//
#include <hip/hip_runtime.h>
#include <hip/hip_bf16.h>

// ---------------------------------------------------------------------------
// Types for CDNA5 WMMA (wave32): v_wmma_f32_16x16x32_bf16
// ---------------------------------------------------------------------------
typedef __bf16 bf16_t;
typedef bf16_t v16bf __attribute__((ext_vector_type(16)));
typedef float  v8f   __attribute__((ext_vector_type(8)));

__device__ __forceinline__ unsigned short f2bf(float f) {
    union { float f; unsigned u; } x; x.f = f;
    unsigned r = x.u + 0x7FFFu + ((x.u >> 16) & 1u);   // round-to-nearest-even
    return (unsigned short)(r >> 16);
}

// Packed f32x2 -> bf16x2 (prefers the HW cvt instruction when declared)
__device__ __forceinline__ unsigned pk2bf(float lo, float hi) {
#if __has_builtin(__builtin_amdgcn_cvt_pk_bf16_f32)
    auto r = __builtin_amdgcn_cvt_pk_bf16_f32(lo, hi);
    union { decltype(r) v; unsigned u; } c; c.v = r;
    return c.u;
#else
    return (unsigned)f2bf(lo) | ((unsigned)f2bf(hi) << 16);
#endif
}

#define BM  128
#define BN  128
#define BK  64
#define LDT 72   // padded LDS K-stride (bf16 elems): 144B rows, 16B aligned
#define SCP 132  // padded score-row stride (floats) for MODE 2 argmin scan

// ---------------------------------------------------------------------------
// Weight pre-pass: W[K,N] f32 -> Wt[N,K] bf16  (32x32 LDS tile transpose)
// ---------------------------------------------------------------------------
__global__ __launch_bounds__(256)
void wconv_transpose(const float* __restrict__ W, unsigned short* __restrict__ Wt,
                     int K, int N)
{
    __shared__ float t[32][33];
    const int tx = threadIdx.x & 31, ty = threadIdx.x >> 5;   // 32 x 8
    const int k0 = blockIdx.y * 32, n0 = blockIdx.x * 32;
    #pragma unroll
    for (int i = 0; i < 4; ++i)
        t[ty + i * 8][tx] = W[(size_t)(k0 + ty + i * 8) * N + n0 + tx];
    __syncthreads();
    #pragma unroll
    for (int i = 0; i < 4; ++i)
        Wt[(size_t)(n0 + ty + i * 8) * K + k0 + tx] = f2bf(t[tx][ty + i * 8]);
}

// ---------------------------------------------------------------------------
// VQ pre-pass: cbt[n,k] = bf16(-2*codebook[n,k]);  cnorm[n] = |codebook[n]|^2
// (codebook is [256,64] which is already the Bt [N,K] layout for the GEMM)
// ---------------------------------------------------------------------------
__global__ __launch_bounds__(256)
void vq_prep(const float* __restrict__ cb, unsigned short* __restrict__ cbt,
             float* __restrict__ cnorm)
{
    const int n = threadIdx.x;                 // 256 codes, 1 block
    float s = 0.f;
    #pragma unroll
    for (int i = 0; i < 16; ++i) {
        float4 v = *(const float4*)&cb[n * 64 + (i << 2)];
        s = __builtin_fmaf(v.x, v.x, s);
        s = __builtin_fmaf(v.y, v.y, s);
        s = __builtin_fmaf(v.z, v.z, s);
        s = __builtin_fmaf(v.w, v.w, s);
        *(uint2*)&cbt[n * 64 + (i << 2)] =
            make_uint2(pk2bf(-2.f * v.x, -2.f * v.y), pk2bf(-2.f * v.z, -2.f * v.w));
    }
    cnorm[n] = s;
}

// ---------------------------------------------------------------------------
// Tiled GEMM:  S[M,N] = A[M,K] @ B[K,N] + bias[N]
//   A: f32 (A_IS_F32) or raw bf16;  Bt: bf16, PRE-TRANSPOSED [N,K]
//   MODE 0: C = bf16(relu(S));  MODE 1: C = f32(S)
//   MODE 2: per-row argmin over this block's BN columns ->
//           Cp as float2 partials [gridDim.x][M] = (minScore, colIndex)
// 256 threads = 8 waves (4M x 2N); wave computes 32x64 = 2x4 WMMA tiles.
// BK=64 (two WMMA K-steps), explicit LDS double buffering.
// ---------------------------------------------------------------------------
template <bool A_IS_F32, int MODE>
__global__ __launch_bounds__(256)
void gemm_bf16_wmma(const void* __restrict__ Ap,
                    const unsigned short* __restrict__ Bt,
                    const float* __restrict__ bias, void* __restrict__ Cp,
                    int M, int N, int K)
{
    // single LDS block: [2 x BM*LDT | 2 x BN*LDT]; MODE 2 reuses it for scores
    __shared__ __align__(16) unsigned short smem[2 * (BM + BN) * LDT];
    unsigned short* sAb = smem;                       // buf stride BM*LDT
    unsigned short* sBb = smem + 2 * BM * LDT;        // buf stride BN*LDT

    const int tid  = threadIdx.x;
    const int bM   = blockIdx.y * BM;
    const int bN   = blockIdx.x * BN;
    const int lane = tid & 31;
    const int w    = tid >> 5;
    const int wm   = w & 3;        // wave M index -> 32 rows
    const int wn   = w >> 2;       // wave N index -> 64 cols
    const int ln   = lane & 15;
    const int lh   = lane >> 4;    // half-wave select

    v8f acc[2][4] = {};

    uint4 rA[4];    // bf16 A staging
    uint2 rAf[8];   // f32  A staging, converted at load
    uint4 rB[4];    // bf16 B staging

    auto loadA = [&](int k0) {
        if (A_IS_F32) {
            const float* A = (const float*)Ap;
            #pragma unroll
            for (int i = 0; i < 8; ++i) {          // 2048 float4 chunks
                int c = tid + i * 256;
                int row = c >> 4, kc = (c & 15) << 2;
                float4 v = *(const float4*)(A + (size_t)(bM + row) * K + k0 + kc);
                rAf[i] = make_uint2(pk2bf(v.x, v.y), pk2bf(v.z, v.w));
            }
        } else {
            const unsigned short* A = (const unsigned short*)Ap;
            #pragma unroll
            for (int i = 0; i < 4; ++i) {          // 1024 uint4 chunks
                int c = tid + i * 256;
                int row = c >> 3, kc = (c & 7) << 3;
                rA[i] = *(const uint4*)(A + (size_t)(bM + row) * K + k0 + kc);
            }
        }
    };
    auto loadB = [&](int k0) {
        #pragma unroll
        for (int i = 0; i < 4; ++i) {              // 1024 uint4 chunks
            int c = tid + i * 256;
            int row = c >> 3, kc = (c & 7) << 3;
            uint4 v = make_uint4(0u, 0u, 0u, 0u);
            if (bN + row < N)
                v = *(const uint4*)(Bt + (size_t)(bN + row) * K + k0 + kc);
            rB[i] = v;
        }
    };
    auto storeT = [&](int buf) {
        if (A_IS_F32) {
            #pragma unroll
            for (int i = 0; i < 8; ++i) {
                int c = tid + i * 256;
                int row = c >> 4, kc = (c & 15) << 2;
                *(uint2*)&sAb[buf * (BM * LDT) + row * LDT + kc] = rAf[i];
            }
        } else {
            #pragma unroll
            for (int i = 0; i < 4; ++i) {
                int c = tid + i * 256;
                int row = c >> 3, kc = (c & 7) << 3;
                *(uint4*)&sAb[buf * (BM * LDT) + row * LDT + kc] = rA[i];
            }
        }
        #pragma unroll
        for (int i = 0; i < 4; ++i) {
            int c = tid + i * 256;
            int row = c >> 3, kc = (c & 7) << 3;
            *(uint4*)&sBb[buf * (BN * LDT) + row * LDT + kc] = rB[i];
        }
    };

    const int nk = K / BK;

    loadA(0); loadB(0);
    storeT(0);
    __syncthreads();

    for (int kt = 0; kt < nk; ++kt) {
        const int cur = kt & 1;

        if (kt + 1 < nk) { loadA((kt + 1) * BK); loadB((kt + 1) * BK); }

        // L2 prefetch of the K+2 weight tile (global_prefetch_b8)
        if (kt + 2 < nk) {
            int pr = tid >> 1, pk = (tid & 1) << 5;
            if (bN + pr < N)
                __builtin_prefetch(Bt + (size_t)(bN + pr) * K + (kt + 2) * BK + pk, 0, 1);
        }

        union Frag { v16bf v; uint4 u[2]; };
        #pragma unroll
        for (int ks = 0; ks < 2; ++ks) {
            const int kb = ks * 32;
            Frag a[2], b[4];
            #pragma unroll
            for (int mi = 0; mi < 2; ++mi) {
                // A 16x32 layout: lanes0-15: K0..7/K16..23 ; lanes16-31: K8..15/K24..31
                int r = wm * 32 + mi * 16 + ln;
                a[mi].u[0] = *(const uint4*)&sAb[cur * (BM * LDT) + r * LDT + kb + lh * 8];
                a[mi].u[1] = *(const uint4*)&sAb[cur * (BM * LDT) + r * LDT + kb + 16 + lh * 8];
            }
            #pragma unroll
            for (int ni = 0; ni < 4; ++ni) {
                // B 32x16 layout: lanes0-15: K0..7/K16..23 ; lanes16-31: K8..15/K24..31
                int n = wn * 64 + ni * 16 + ln;
                b[ni].u[0] = *(const uint4*)&sBb[cur * (BN * LDT) + n * LDT + kb + lh * 8];
                b[ni].u[1] = *(const uint4*)&sBb[cur * (BN * LDT) + n * LDT + kb + 16 + lh * 8];
            }
            #pragma unroll
            for (int mi = 0; mi < 2; ++mi)
                #pragma unroll
                for (int ni = 0; ni < 4; ++ni)
                    acc[mi][ni] = __builtin_amdgcn_wmma_f32_16x16x32_bf16(
                        false, a[mi].v, false, b[ni].v,
                        (short)0, acc[mi][ni], false, false);
        }

        if (kt + 1 < nk) storeT(cur ^ 1);
        __syncthreads();
    }

    // ---- epilogue ----
    // C/D layout: VGPR r: lanes 0-15 -> M=r, lanes 16-31 -> M=r+8; N=lane%16
    if (MODE == 2) {
        // dump score tile to LDS (staging buffers are dead after final barrier)
        float* ssc = (float*)smem;                 // BM x SCP floats (67.5KB <= 73.7KB)
        #pragma unroll
        for (int ni = 0; ni < 4; ++ni) {
            int colt = wn * 64 + ni * 16 + ln;
            float bv = bias[bN + colt];
            #pragma unroll
            for (int mi = 0; mi < 2; ++mi) {
                int rbase = wm * 32 + mi * 16 + lh * 8;
                #pragma unroll
                for (int r = 0; r < 8; ++r)
                    ssc[(rbase + r) * SCP + colt] = acc[mi][ni][r] + bv;
            }
        }
        __syncthreads();
        if (tid < BM) {
            float best = 3.402823466e38f; int bj = 0;
            #pragma unroll 4
            for (int j = 0; j < BN; j += 4) {      // strict < keeps first (argmin)
                float4 v = *(const float4*)&ssc[tid * SCP + j];
                if (v.x < best) { best = v.x; bj = j; }
                if (v.y < best) { best = v.y; bj = j + 1; }
                if (v.z < best) { best = v.z; bj = j + 2; }
                if (v.w < best) { best = v.w; bj = j + 3; }
            }
            float2* part = (float2*)Cp;            // [gridDim.x][M]
            part[(size_t)blockIdx.x * M + bM + tid] = make_float2(best, (float)(bN + bj));
        }
        return;
    }

    #pragma unroll
    for (int ni = 0; ni < 4; ++ni) {
        int col = bN + wn * 64 + ni * 16 + ln;
        if (col >= N) continue;
        float bv = bias[col];
        #pragma unroll
        for (int mi = 0; mi < 2; ++mi) {
            int rbase = bM + wm * 32 + mi * 16 + lh * 8;
            #pragma unroll
            for (int r = 0; r < 8; ++r) {
                float x = acc[mi][ni][r] + bv;
                if (MODE == 0) x = x > 0.f ? x : 0.f;    // fused ReLU
                if (MODE == 1)
                    ((float*)Cp)[(size_t)(rbase + r) * N + col] = x;
                else
                    ((unsigned short*)Cp)[(size_t)(rbase + r) * N + col] = f2bf(x);
            }
        }
    }
}

// ---------------------------------------------------------------------------
// VQ finalize: combine the 2 column-block partials per row (tie -> lower index,
// matching argmin), gather codebook[best] and write f32 + bf16 role_emb.
// ---------------------------------------------------------------------------
__global__ __launch_bounds__(256)
void vq_gather(const float2* __restrict__ part, const float* __restrict__ cb,
               float* __restrict__ out, unsigned short* __restrict__ out_bf, int M)
{
    __shared__ __align__(16) float lcb[256 * 64];
    const int tid = threadIdx.x;
    #pragma unroll
    for (int i = 0; i < 16; ++i) {
        int idx = (tid + i * 256) << 2;
        *(float4*)&lcb[idx] = *(const float4*)&cb[idx];
    }
    __syncthreads();

    const size_t row = (size_t)blockIdx.x * 256 + tid;
    const float2 p0 = part[row];
    const float2 p1 = part[(size_t)M + row];
    const int bi = (p1.x < p0.x) ? (int)p1.y : (int)p0.y;

    const float* cp = &lcb[bi * 64];
    #pragma unroll
    for (int i = 0; i < 16; ++i) {
        float4 cv = *(const float4*)&cp[i << 2];
        *(float4*)&out[row * 64 + (i << 2)] = cv;
        *(uint2*)&out_bf[row * 64 + (i << 2)] =
            make_uint2(pk2bf(cv.x, cv.y), pk2bf(cv.z, cv.w));
    }
}

// ---------------------------------------------------------------------------
// Host-side orchestration
// ---------------------------------------------------------------------------
extern "C" void kernel_launch(void* const* d_in, const int* in_sizes, int n_in,
                              void* d_out, int out_size, void* d_ws, size_t ws_size,
                              hipStream_t stream) {
    (void)in_sizes; (void)n_in; (void)out_size; (void)ws_size;

    const float* inputs   = (const float*)d_in[0];   // [131072, 256]
    const float* W0       = (const float*)d_in[1];   // [256, 512]
    const float* b0       = (const float*)d_in[2];
    const float* W1       = (const float*)d_in[3];   // [512, 512]
    const float* b1       = (const float*)d_in[4];
    const float* W2       = (const float*)d_in[5];   // [512, 64]
    const float* b2       = (const float*)d_in[6];
    const float* codebook = (const float*)d_in[7];   // [256, 64]
    const float* W3       = (const float*)d_in[8];   // [64, 512]
    const float* b3       = (const float*)d_in[9];
    const float* W4       = (const float*)d_in[10];  // [4096, 512]
    const float* b4       = (const float*)d_in[11];
    const float* W5       = (const float*)d_in[12];  // [512, 512]
    const float* b5       = (const float*)d_in[13];

    const int NROWS = 131072, NG = 16384;            // NG = NROWS / 8 agents
    float* recon = (float*)d_out;                    // [16384, 512]
    float* z_e   = (float*)d_out + (size_t)NG * 512; // [131072, 64]
    float* role  = z_e + (size_t)NROWS * 64;         // [131072, 64]

    // workspace carve-up (units: ushort; every region has 16B-multiple size)
    size_t off = 0;
    auto carve = [&](size_t nelem) {
        unsigned short* p = (unsigned short*)d_ws + off; off += nelem; return p;
    };
    unsigned short* ws0   = carve((size_t)NROWS * 512);   // activations ping (bf16)
    unsigned short* ws1   = carve((size_t)NROWS * 512);   // activations pong (bf16)
    unsigned short* wt0   = carve((size_t)256 * 512);     // W0^T  bf16 [512,256]
    unsigned short* wt1   = carve((size_t)512 * 512);     // W1^T
    unsigned short* wt2   = carve((size_t)512 * 64);      // W2^T  [64,512]
    unsigned short* wt3   = carve((size_t)64 * 512);      // W3^T  [512,64]
    unsigned short* wt4   = carve((size_t)4096 * 512);    // W4^T  [512,4096]
    unsigned short* wt5   = carve((size_t)512 * 512);     // W5^T
    unsigned short* roleb = carve((size_t)NROWS * 64);    // role_emb bf16
    unsigned short* cbt   = carve((size_t)256 * 64);      // -2*codebook bf16 [256,64]
    float*  cnorm = (float*)carve((size_t)256 * 2);       // |c|^2 per code
    float2* part  = (float2*)carve((size_t)2 * NROWS * 4);// argmin partials [2][M]

    dim3 blk(256);
    auto grd  = [](int M, int N) { return dim3((N + BN - 1) / BN, (M + BM - 1) / BM); };
    auto grdT = [](int K, int N) { return dim3(N / 32, K / 32); };

    // ---- one-time weight convert + transpose to bf16 [N,K]; VQ tables ----
    wconv_transpose<<<grdT(256, 512),  blk, 0, stream>>>(W0, wt0, 256, 512);
    wconv_transpose<<<grdT(512, 512),  blk, 0, stream>>>(W1, wt1, 512, 512);
    wconv_transpose<<<grdT(512, 64),   blk, 0, stream>>>(W2, wt2, 512, 64);
    wconv_transpose<<<grdT(64, 512),   blk, 0, stream>>>(W3, wt3, 64, 512);
    wconv_transpose<<<grdT(4096, 512), blk, 0, stream>>>(W4, wt4, 4096, 512);
    wconv_transpose<<<grdT(512, 512),  blk, 0, stream>>>(W5, wt5, 512, 512);
    vq_prep<<<1, 256, 0, stream>>>(codebook, cbt, cnorm);

    // ---- pipeline ----
    // h0 = relu(inputs @ W0 + b0)                      f32 in -> bf16 out
    gemm_bf16_wmma<true,  0><<<grd(NROWS, 512), blk, 0, stream>>>(inputs, wt0, b0, ws0, NROWS, 512, 256);
    // h1 = relu(h0 @ W1 + b1)                          bf16 -> bf16
    gemm_bf16_wmma<false, 0><<<grd(NROWS, 512), blk, 0, stream>>>(ws0, wt1, b1, ws1, NROWS, 512, 512);
    // z_e = h1 @ W2 + b2                               bf16 -> f32 (output)
    gemm_bf16_wmma<false, 1><<<grd(NROWS, 64),  blk, 0, stream>>>(ws1, wt2, b2, z_e, NROWS, 64, 512);
    // VQ distances as WMMA GEMM: score = z_e @ (-2 cb^T) + |c|^2, per-block argmin
    gemm_bf16_wmma<true,  2><<<grd(NROWS, 256), blk, 0, stream>>>(z_e, cbt, cnorm, part, NROWS, 256, 64);
    // role_emb = codebook[argmin]; forward z_q == role_emb
    vq_gather<<<NROWS / 256, blk, 0, stream>>>(part, codebook, role, roleb, NROWS);
    // rec = relu(z_q @ W3 + b3)                        bf16 -> bf16
    gemm_bf16_wmma<false, 0><<<grd(NROWS, 512), blk, 0, stream>>>(roleb, wt3, b3, ws0, NROWS, 512, 64);
    // rec2 = relu(rec.reshape(16384,4096) @ W4 + b4)   bf16 -> bf16
    gemm_bf16_wmma<false, 0><<<grd(NG, 512),    blk, 0, stream>>>(ws0, wt4, b4, ws1, NG, 512, 4096);
    // recon = rec2 @ W5 + b5                           bf16 -> f32 (output)
    gemm_bf16_wmma<false, 1><<<grd(NG, 512),    blk, 0, stream>>>(ws1, wt5, b5, recon, NG, 512, 512);
}
